// Q_Attention_360777253090
// MI455X (gfx1250) — 8-bit matrix operand order repaired and checked by reading, compiled, not run on hardware
//
#include <hip/hip_runtime.h>
#include <math.h>


typedef __attribute__((ext_vector_type(8))) int v8i;
typedef __attribute__((ext_vector_type(4))) int vi4;

union F8 {
  v8i v;
  long long l[4];
  vi4 q[2];
  int i[8];
};

#define B_   32
#define N_   512
#define C_   768
#define H_   12
#define HD_  64
#define M_   (B_ * N_)
#define NQKV 2304

__device__ __forceinline__ float clampf(float x, float lo, float hi) {
  return fminf(fmaxf(x, lo), hi);
}

__device__ __forceinline__ void async_b128(unsigned lds_off, const void* gaddr) {
  asm volatile("global_load_async_to_lds_b128 %0, %1, off"
               :: "v"(lds_off), "v"(gaddr)
               : "memory");
}
__device__ __forceinline__ void wait_async0() {
  asm volatile("s_wait_asynccnt 0" ::: "memory");
}
__device__ __forceinline__ unsigned lds_off_of(const void* p) {
  return (unsigned)(unsigned long long)p;
}

__global__ void means_kernel(const float* qkv_act_alpha, const float* qact,
                             const float* kact, const float* vact,
                             const float* attnact, const float* proj_act_alpha,
                             float* scal) {
  if (threadIdx.x == 0 && blockIdx.x == 0) {
    float s = 0.f;
    for (int i = 0; i < C_; ++i) s += qkv_act_alpha[i];
    scal[0] = s / (float)C_;
    s = 0.f; for (int i = 0; i < H_; ++i) s += qact[i];    scal[1] = s / (float)H_;
    s = 0.f; for (int i = 0; i < H_; ++i) s += kact[i];    scal[2] = s / (float)H_;
    s = 0.f; for (int i = 0; i < H_; ++i) s += vact[i];    scal[3] = s / (float)H_;
    s = 0.f; for (int i = 0; i < H_; ++i) s += attnact[i]; scal[4] = s / (float)H_;
    s = 0.f; for (int i = 0; i < C_; ++i) s += proj_act_alpha[i];
    scal[5] = s / (float)C_;
  }
}

__global__ void quant_w_kernel(const float* qkv_w, const float* qkv_alpha,
                               const float* proj_w, const float* proj_alpha,
                               signed char* wq, signed char* pwq) {
  int idx = blockIdx.x * blockDim.x + threadIdx.x;
  const int nq = NQKV * C_;
  if (idx < nq) {
    int r = idx / C_;
    float v = clampf(qkv_w[idx] / qkv_alpha[r], -8.f, 7.f);
    wq[idx] = (signed char)(int)rintf(v);
  } else if (idx < nq + C_ * C_) {
    int j = idx - nq;
    int r = j / C_;
    float v = clampf(proj_w[j] / proj_alpha[r], -8.f, 7.f);
    pwq[j] = (signed char)(int)rintf(v);
  }
}

__global__ void quant_x_bias_kernel(const float* x0, const float* qkv_bias,
                                    const float* qkv_alpha, const float* scal,
                                    signed char* x0q, int* biasi) {
  int idx = blockIdx.x * blockDim.x + threadIdx.x;
  float a_in = scal[0];
  if (idx < M_ * C_) {
    float v = clampf(x0[idx] / a_in, -8.f, 7.f);
    x0q[idx] = (signed char)(int)rintf(v);
  }
  if (idx < NQKV) {
    biasi[idx] = (int)truncf(qkv_bias[idx] / (a_in * qkv_alpha[idx]));
  }
}

__global__ __launch_bounds__(256) void qkv_gemm_kernel(
    const signed char* __restrict__ x0q, const signed char* __restrict__ wq,
    const int* __restrict__ biasi, const float* __restrict__ qkv_alpha,
    const float* __restrict__ scal, int* __restrict__ q0i,
    int* __restrict__ k0i, signed char* __restrict__ v2T) {
  extern __shared__ unsigned char dyn_smem[];
  signed char* sA = (signed char*)dyn_smem;
  const int tid  = threadIdx.x;
  const int lane = tid & 31;
  const int wave = tid >> 5;
  const int wm = wave & 1, wn = wave >> 1;
  const int Mbase = blockIdx.x * 32 + wm * 16;
  const int Nbase = blockIdx.y * 256 + wn * 64;
  const int half = (lane >> 4) & 1;
  const int lr   = lane & 15;

  {
    const signed char* gA = x0q + (size_t)(blockIdx.x * 32) * C_;
#pragma unroll
    for (int i = 0; i < 6; ++i) {
      int off = i * 4096 + tid * 16;
      async_b128(lds_off_of(sA + off), gA + off);
    }
    wait_async0();
    __syncthreads();
  }

  v8i zero = {0, 0, 0, 0, 0, 0, 0, 0};
  v8i acc[4] = {zero, zero, zero, zero};

  const int arowL = wm * 16 + lr;
  for (int k0 = 0; k0 < C_; k0 += 64) {
    F8 a;
    const signed char* ap = sA + arowL * C_ + k0 + half * 16;
    a.l[0] = *(const long long*)(ap);
    a.l[1] = *(const long long*)(ap + 8);
    a.l[2] = *(const long long*)(ap + 32);
    a.l[3] = *(const long long*)(ap + 40);
#pragma unroll
    for (int nt = 0; nt < 4; ++nt) {
      int col = Nbase + nt * 16 + lr;
      F8 b;
      const signed char* bp = wq + col * C_ + k0 + half * 16;
      b.q[0] = *(const vi4*)(bp);
      b.q[1] = *(const vi4*)(bp + 32);
      acc[nt] = __builtin_amdgcn_wmma_i32_16x16x64_iu8(
          true, a.v, true, b.v, acc[nt], false, false);
    }
  }

  const float a_in = scal[0];
  const float va   = scal[3];
#pragma unroll
  for (int nt = 0; nt < 4; ++nt) {
    int col   = Nbase + nt * 16 + lr;
    int bi    = biasi[col];
    int three = col / C_;
    int rem   = col - three * C_;
    int h     = rem >> 6;
    int d     = rem & 63;
    float vs  = a_in * qkv_alpha[col] / va;
    F8 cu;
    cu.v = acc[nt];
#pragma unroll
    for (int r = 0; r < 8; ++r) {
      int tokenRow = Mbase + r + half * 8;
      int b = tokenRow >> 9;
      int n = tokenRow & 511;
      int bh = b * H_ + h;
      int val = cu.i[r] + bi;
      if (three == 0) {
        q0i[(bh * N_ + n) * HD_ + d] = val;
      } else if (three == 1) {
        k0i[(bh * N_ + n) * HD_ + d] = val;
      } else {
        float v = clampf((float)val * vs, -4.f, 3.f);
        v2T[(bh * HD_ + d) * N_ + n] = (signed char)(int)rintf(v);
      }
    }
  }
}

__global__ __launch_bounds__(64) void norm_quant_kernel(
    const int* __restrict__ qin, const int* __restrict__ kin,
    const float* __restrict__ qkv_alpha, const float* __restrict__ nqw,
    const float* __restrict__ nqb, const float* __restrict__ nkw,
    const float* __restrict__ nkb, const float* __restrict__ scal,
    signed char* __restrict__ q2, signed char* __restrict__ k2) {
  int row   = blockIdx.x;
  int which = blockIdx.y;
  int d     = threadIdx.x;
  const int* src = which ? kin : qin;
  int h = (row >> 9) % H_;
  float alpha = qkv_alpha[which * C_ + h * HD_ + d];
  float x = (float)src[row * HD_ + d] * alpha;

  __shared__ float s1[64];
  __shared__ float s2[64];
  s1[d] = x;
  s2[d] = x * x;
  __syncthreads();
  for (int s = 32; s > 0; s >>= 1) {
    if (d < s) {
      s1[d] += s1[d + s];
      s2[d] += s2[d + s];
    }
    __syncthreads();
  }
  float mean = s1[0] * (1.f / 64.f);
  float var  = (s2[0] - 64.f * mean * mean) * (1.f / 63.f);
  float sd   = sqrtf(fmaxf(var, 0.f)) + 1e-5f;
  float xn   = (x - mean) / sd;
  float nw   = which ? nkw[d] : nqw[d];
  float nb   = which ? nkb[d] : nqb[d];
  float act  = which ? scal[2] : scal[1];
  float y = clampf((xn * nw + nb) / act, -4.f, 3.f);
  signed char* dst = which ? k2 : q2;
  dst[row * HD_ + d] = (signed char)(int)rintf(y);
}

__global__ __launch_bounds__(128) void attn_kernel(
    const signed char* __restrict__ q2, const signed char* __restrict__ k2,
    const signed char* __restrict__ v2T, const float* __restrict__ scal,
    signed char* __restrict__ x1q) {
  extern __shared__ unsigned char dyn_smem[];
  signed char* sK = (signed char*)dyn_smem;
  signed char* sV = (signed char*)(dyn_smem + 32768);
  unsigned char* sAttn = dyn_smem + 65536;

  const int tid  = threadIdx.x;
  const int lane = tid & 31;
  const int wave = tid >> 5;
  const int half = (lane >> 4) & 1;
  const int lr   = lane & 15;
  const int bh   = blockIdx.x;
  const int qBase = blockIdx.y * 64 + wave * 16;
  const int b = bh / H_;
  const int h = bh - b * H_;

  {
    const signed char* gk = k2 + (size_t)bh * N_ * HD_;
    const signed char* gv = v2T + (size_t)bh * HD_ * N_;
#pragma unroll
    for (int i = 0; i < 16; ++i) {
      int off = i * 2048 + tid * 16;
      async_b128(lds_off_of(sK + off), gk + off);
      async_b128(lds_off_of(sV + off), gv + off);
    }
    wait_async0();
    __syncthreads();
  }

  const float qa = scal[1], ka = scal[2], va = scal[3], aa = scal[4],
              pa = scal[5];
  const float C1x    = 1.44269504088896340736f * 0.125f * qa * ka * 128.0f;
  const float inv128 = 1.0f / 128.0f;

  F8 qf;
  {
    const signed char* ap = q2 + (bh * N_ + qBase + lr) * HD_ + half * 16;
    qf.l[0] = *(const long long*)(ap);
    qf.l[1] = *(const long long*)(ap + 8);
    qf.l[2] = *(const long long*)(ap + 32);
    qf.l[3] = *(const long long*)(ap + 40);
  }

  float rs[8];
#pragma unroll
  for (int r = 0; r < 8; ++r) rs[r] = 0.f;

  for (int nt = 0; nt < 32; ++nt) {
    int key = nt * 16 + lr;
    F8 bf;
    const signed char* bp = sK + key * HD_ + half * 16;
    bf.q[0] = *(const vi4*)(bp);
    bf.q[1] = *(const vi4*)(bp + 32);
    v8i z = {0, 0, 0, 0, 0, 0, 0, 0};
    F8 cu;
    cu.v = __builtin_amdgcn_wmma_i32_16x16x64_iu8(true, qf.v, true, bf.v, z,
                                                  false, false);
#pragma unroll
    for (int r = 0; r < 8; ++r) {
      float e = truncf(C1x * (float)cu.i[r]);
      rs[r] += exp2f(rintf(e * inv128));
    }
  }
#pragma unroll
  for (int r = 0; r < 8; ++r) {
    float v = rs[r];
    v += __shfl_xor(v, 1);
    v += __shfl_xor(v, 2);
    v += __shfl_xor(v, 4);
    v += __shfl_xor(v, 8);
    rs[r] = v;
  }

  v8i zero = {0, 0, 0, 0, 0, 0, 0, 0};
  v8i accv[4] = {zero, zero, zero, zero};

  for (int ch = 0; ch < 8; ++ch) {
#pragma unroll
    for (int sub = 0; sub < 4; ++sub) {
      int nt  = ch * 4 + sub;
      int key = nt * 16 + lr;
      F8 bf;
      const signed char* bp = sK + key * HD_ + half * 16;
      bf.q[0] = *(const vi4*)(bp);
      bf.q[1] = *(const vi4*)(bp + 32);
      v8i z = {0, 0, 0, 0, 0, 0, 0, 0};
      F8 cu;
      cu.v = __builtin_amdgcn_wmma_i32_16x16x64_iu8(true, qf.v, true, bf.v, z,
                                                    false, false);
#pragma unroll
      for (int r = 0; r < 8; ++r) {
        float e  = truncf(C1x * (float)cu.i[r]);
        float ev = exp2f(rintf(e * inv128));
        float v  = clampf(ev / (rs[r] * aa), 0.f, 7.f);
        sAttn[((wave * 16) + r + half * 8) * 64 + sub * 16 + lr] =
            (unsigned char)(int)rintf(v);
      }
    }
    __syncthreads();
    F8 af;
    {
      const unsigned char* ap = sAttn + (wave * 16 + lr) * 64 + half * 16;
      af.l[0] = *(const long long*)(ap);
      af.l[1] = *(const long long*)(ap + 8);
      af.l[2] = *(const long long*)(ap + 32);
      af.l[3] = *(const long long*)(ap + 40);
    }
#pragma unroll
    for (int dt = 0; dt < 4; ++dt) {
      int dcol = dt * 16 + lr;
      F8 bf;
      const signed char* bp = sV + dcol * N_ + ch * 64 + half * 16;
      bf.q[0] = *(const vi4*)(bp);
      bf.q[1] = *(const vi4*)(bp + 32);
      accv[dt] = __builtin_amdgcn_wmma_i32_16x16x64_iu8(
          false  , af.v, true  , bf.v, accv[dt],
          false, false);
    }
    __syncthreads();
  }

  const float s2 = aa * va / pa;
#pragma unroll
  for (int dt = 0; dt < 4; ++dt) {
    int dcol = dt * 16 + lr;
    F8 cu;
    cu.v = accv[dt];
#pragma unroll
    for (int r = 0; r < 8; ++r) {
      int n = qBase + r + half * 8;
      float v = clampf((float)cu.i[r] * s2, -4.f, 3.f);
      x1q[(b * N_ + n) * C_ + h * HD_ + dcol] = (signed char)(int)rintf(v);
    }
  }
}

__global__ __launch_bounds__(256) void proj_gemm_kernel(
    const signed char* __restrict__ x1q, const signed char* __restrict__ pwq,
    const float* __restrict__ proj_alpha, const float* __restrict__ proj_bias,
    const float* __restrict__ scal, float* __restrict__ out) {
  extern __shared__ unsigned char dyn_smem[];
  signed char* sA = (signed char*)dyn_smem;
  const int tid  = threadIdx.x;
  const int lane = tid & 31;
  const int wave = tid >> 5;
  const int wm = wave & 1, wn = wave >> 1;
  const int Mbase = blockIdx.x * 32 + wm * 16;
  const int Nbase = blockIdx.y * 256 + wn * 64;
  const int half = (lane >> 4) & 1;
  const int lr   = lane & 15;

  {
    const signed char* gA = x1q + (size_t)(blockIdx.x * 32) * C_;
#pragma unroll
    for (int i = 0; i < 6; ++i) {
      int off = i * 4096 + tid * 16;
      async_b128(lds_off_of(sA + off), gA + off);
    }
    wait_async0();
    __syncthreads();
  }

  v8i zero = {0, 0, 0, 0, 0, 0, 0, 0};
  v8i acc[4] = {zero, zero, zero, zero};

  const int arowL = wm * 16 + lr;
  for (int k0 = 0; k0 < C_; k0 += 64) {
    F8 a;
    const signed char* ap = sA + arowL * C_ + k0 + half * 16;
    a.l[0] = *(const long long*)(ap);
    a.l[1] = *(const long long*)(ap + 8);
    a.l[2] = *(const long long*)(ap + 32);
    a.l[3] = *(const long long*)(ap + 40);
#pragma unroll
    for (int nt = 0; nt < 4; ++nt) {
      int col = Nbase + nt * 16 + lr;
      F8 b;
      const signed char* bp = pwq + col * C_ + k0 + half * 16;
      b.q[0] = *(const vi4*)(bp);
      b.q[1] = *(const vi4*)(bp + 32);
      acc[nt] = __builtin_amdgcn_wmma_i32_16x16x64_iu8(
          true, a.v, true, b.v, acc[nt], false, false);
    }
  }

  const float pa = scal[5];
#pragma unroll
  for (int nt = 0; nt < 4; ++nt) {
    int col = Nbase + nt * 16 + lr;
    float sc = proj_alpha[col] * pa;
    float bs = proj_bias[col];
    F8 cu;
    cu.v = acc[nt];
#pragma unroll
    for (int r = 0; r < 8; ++r) {
      int tokenRow = Mbase + r + half * 8;
      out[tokenRow * C_ + col] = (float)cu.i[r] * sc + bs;
    }
  }
}

extern "C" void kernel_launch(void* const* d_in, const int* in_sizes, int n_in,
                              void* d_out, int out_size, void* d_ws,
                              size_t ws_size, hipStream_t stream) {
  const float* x0             = (const float*)d_in[0];
  const float* qkv_w          = (const float*)d_in[1];
  const float* qkv_alpha      = (const float*)d_in[2];
  const float* qkv_bias       = (const float*)d_in[3];
  const float* qkv_act_alpha  = (const float*)d_in[4];
  const float* proj_w         = (const float*)d_in[5];
  const float* proj_alpha     = (const float*)d_in[6];
  const float* proj_bias      = (const float*)d_in[7];
  const float* proj_act_alpha = (const float*)d_in[8];
  const float* normq_w        = (const float*)d_in[9];
  const float* normq_b        = (const float*)d_in[10];
  const float* normk_w        = (const float*)d_in[11];
  const float* normk_b        = (const float*)d_in[12];
  const float* qact_alpha     = (const float*)d_in[13];
  const float* kact_alpha     = (const float*)d_in[14];
  const float* vact_alpha     = (const float*)d_in[15];
  const float* attnact_alpha  = (const float*)d_in[16];
  float* out = (float*)d_out;

  size_t off = 0;
  auto carve = [&](size_t bytes) -> char* {
    char* p = (char*)d_ws + off;
    off = (off + bytes + 255) & ~(size_t)255;
    return p;
  };
  float*       scal  = (float*)carve(64 * sizeof(float));
  signed char* wq    = (signed char*)carve((size_t)NQKV * C_);
  signed char* pwq   = (signed char*)carve((size_t)C_ * C_);
  int*         biasi = (int*)carve((size_t)NQKV * sizeof(int));
  signed char* x0q   = (signed char*)carve((size_t)M_ * C_);
  int*         q0i   = (int*)carve((size_t)B_ * H_ * N_ * HD_ * sizeof(int));
  int*         k0i   = (int*)carve((size_t)B_ * H_ * N_ * HD_ * sizeof(int));
  signed char* v2T   = (signed char*)carve((size_t)B_ * H_ * HD_ * N_);
  signed char* q2    = (signed char*)carve((size_t)B_ * H_ * N_ * HD_);
  signed char* k2    = (signed char*)carve((size_t)B_ * H_ * N_ * HD_);
  signed char* x1q   = (signed char*)carve((size_t)M_ * C_);
  (void)ws_size;

  means_kernel<<<1, 64, 0, stream>>>(qkv_act_alpha, qact_alpha, kact_alpha,
                                     vact_alpha, attnact_alpha, proj_act_alpha,
                                     scal);

  {
    int total = NQKV * C_ + C_ * C_;
    quant_w_kernel<<<(total + 255) / 256, 256, 0, stream>>>(
        qkv_w, qkv_alpha, proj_w, proj_alpha, wq, pwq);
  }

  {
    int total = M_ * C_;
    quant_x_bias_kernel<<<(total + 255) / 256, 256, 0, stream>>>(
        x0, qkv_bias, qkv_alpha, scal, x0q, biasi);
  }

  qkv_gemm_kernel<<<dim3(M_ / 32, NQKV / 256), 256, 24576, stream>>>(
      x0q, wq, biasi, qkv_alpha, scal, q0i, k0i, v2T);

  norm_quant_kernel<<<dim3(B_ * H_ * N_, 2), 64, 0, stream>>>(
      q0i, k0i, qkv_alpha, normq_w, normq_b, normk_w, normk_b, scal, q2, k2);

  attn_kernel<<<dim3(B_ * H_, N_ / 64), 128, 69632, stream>>>(q2, k2, v2T,
                                                              scal, x1q);

  proj_gemm_kernel<<<dim3(M_ / 32, C_ / 256), 256, 24576, stream>>>(
      x1q, pwq, proj_alpha, proj_bias, scal, out);
}
